// STODEIntegrator_55722905698536
// MI455X (gfx1250) — matrix operand order repaired and checked by reading, not compiled, not run on hardware
//
#include <hip/hip_runtime.h>
#include <hip/hip_bf16.h>
#include <math.h>

// ---------------------------------------------------------------------------
// STODE integrator for MI455X (gfx1250).
//  - Row-wise independent dynamics -> persistent block per 16-row stripe.
//  - State fp32 in LDS (+ bf16 mirror for WMMA A operands).
//  - GEMMs via V_WMMA_F32_16X16X32_BF16 (f32 accumulate); weights bf16 in L2.
//  - Activations via native v_tanh_f32 (TRANS pipe, co-executes with WMMA).
// ---------------------------------------------------------------------------

#define HID     256
#define NROWS   2048
#define BATCH   8
#define NSTEPS  10
#define ROWS    16          // rows per block (one WMMA M-tile)
#define THREADS 128         // 4 wave32
#define NWAVES  4

typedef __attribute__((ext_vector_type(16))) __bf16 v16bf;
typedef __attribute__((ext_vector_type(8)))  float  v8f;

union BF16Frag {
    v16bf          v;
    unsigned short u[16];
    uint4          q[2];
};

__device__ __forceinline__ unsigned short f2bf(float x) {
    unsigned int b = __builtin_bit_cast(unsigned int, x);
    unsigned int r = b + 0x7FFFu + ((b >> 16) & 1u);   // round-to-nearest-even
    return (unsigned short)(r >> 16);
}

__device__ __forceinline__ float fast_tanh(float x) {
#if __has_builtin(__builtin_amdgcn_tanhf)
    return __builtin_amdgcn_tanhf(x);       // v_tanh_f32 (gfx1250 TRANS op)
#else
    return tanhf(x);
#endif
}

__device__ __forceinline__ float sigmoid_f(float x) {
    return 0.5f * (1.0f + fast_tanh(0.5f * x));   // single TRANS op
}

// ---------------------------------------------------------------------------
// Tiled GEMM: for each output tile, store( A[16 x KC*32] * W^T + bias )
//   As  : bf16 in LDS, row-major ushort, leading dim lda (A fragments are
//         two ds_load_b128 per K-chunk, no conversion VALU)
//   W   : bf16 in global memory (L2-hot), [ntiles*16 x ldw] row-major
//   store(v, m, col): applies activation/fusion and writes result
// ---------------------------------------------------------------------------
template <int KC, typename Store>
__device__ __forceinline__ void mm_tiles(const unsigned short* As, int lda,
                                         const unsigned short* __restrict__ W,
                                         int ldw,
                                         const float* __restrict__ bias,
                                         int ntiles, Store store)
{
    const int lane = threadIdx.x & 31;
    const int wave = threadIdx.x >> 5;
    const int nl   = lane & 15;   // column within tile / A row
    const int hi   = lane >> 4;   // K-half selector

    // A fragments: lane (nl,hi) -> row nl, K = kc*32 + 8*hi + {0..7, 16..23}
    BF16Frag afr[KC];
#pragma unroll
    for (int kc = 0; kc < KC; ++kc) {
        const unsigned short* ap = As + nl * lda + kc * 32 + 8 * hi;
        afr[kc].q[0] = *(const uint4*)(ap);
        afr[kc].q[1] = *(const uint4*)(ap + 16);
    }

    for (int nt = wave; nt < ntiles; nt += NWAVES) {
        const int col = nt * 16 + nl;          // output column == W row
        v8f acc = {};
        // B fragment: lane holds W[col][kc*32 + 8*hi + {0..7, 16..23}]
        const uint4* wp = (const uint4*)(W + (size_t)col * ldw + hi * 8);
#pragma unroll
        for (int kc = 0; kc < KC; ++kc) {
            BF16Frag bf;
            bf.q[0] = wp[kc * 4 + 0];
            bf.q[1] = wp[kc * 4 + 2];
            acc = __builtin_amdgcn_wmma_f32_16x16x32_bf16(
                      false, afr[kc].v, false, bf.v, (short)0, acc, false, false);
        }
        const float bcol = bias[col];
#pragma unroll
        for (int r = 0; r < 8; ++r) {
            const int m = 8 * hi + r;          // C/D layout: M = r + 8*hi
            store(acc[r] + bcol, m, col);
        }
    }
}

struct Params {
    const unsigned short *Wg, *Wc, *Ws1, *Ws2, *Wf;   // bf16 weights (d_ws)
    const float *bg, *bc, *wt, *b1, *b2, *bf;         // fp32 biases / Wt_w
};

// One evaluation of _dynamics: stage (f32 stF + bf16 stB) -> dstK (f32).
// dstK may alias stF.
__device__ void dynamics(float t, const float* stF, const unsigned short* stB,
                         float* dstK, float* sG, unsigned short* sAb,
                         unsigned short* sCb, const Params& p)
{
    // gates = sigmoid(H @ Wg^T + bg)            [16 x 512] fp32
    {
        float* g = sG;
        mm_tiles<8>(stB, HID, p.Wg, HID, p.bg, 32,
                    [=](float v, int m, int col) {
                        g[m * 512 + col] = sigmoid_f(v);
                    });
    }
    __syncthreads();

    // sAb = bf16(r * H)   (r = gates[:, 256:])
    for (int i = threadIdx.x; i < ROWS * HID; i += THREADS) {
        const int m = i >> 8, k = i & 255;
        sAb[i] = f2bf(sG[m * 512 + 256 + k] * stF[i]);
    }
    __syncthreads();

    // h_tilde = tanh((r*H) @ Wc^T + bc) + tanh(t*Wt);  dh_temp -> sCb[:, :256]
    {
        const float* wt = p.wt;
        const float* sg = sG;
        const float* st = stF;
        unsigned short* c = sCb;
        const float tt = t;
        mm_tiles<8>(sAb, HID, p.Wc, HID, p.bc, 16,
                    [=](float v, int m, int col) {
                        const float ht = fast_tanh(v) + fast_tanh(tt * wt[col]);
                        const float z  = sg[m * 512 + col];
                        c[m * 512 + col] = f2bf((1.0f - z) * (ht - st[m * HID + col]));
                    });
    }
    __syncthreads();

    // s1 = tanh(H @ Ws1^T + b1) -> sAb (bf16)
    {
        unsigned short* a = sAb;
        mm_tiles<8>(stB, HID, p.Ws1, HID, p.b1, 16,
                    [=](float v, int m, int col) {
                        a[m * HID + col] = f2bf(fast_tanh(v));
                    });
    }
    __syncthreads();

    // dh_spat = s1 @ Ws2^T + b2 -> sCb[:, 256:] (bf16)
    {
        unsigned short* c = sCb;
        mm_tiles<8>(sAb, HID, p.Ws2, HID, p.b2, 16,
                    [=](float v, int m, int col) {
                        c[m * 512 + 256 + col] = f2bf(v);
                    });
    }
    __syncthreads();

    // out = tanh(comb @ Wf^T + bf) -> dstK (f32)
    {
        float* d = dstK;
        mm_tiles<16>(sCb, 2 * HID, p.Wf, 2 * HID, p.bf, 16,
                     [=](float v, int m, int col) {
                         d[m * HID + col] = fast_tanh(v);
                     });
    }
    __syncthreads();
}

__global__ void __launch_bounds__(THREADS)
stode_kernel(const float* __restrict__ h0, const float* __restrict__ tsp,
             Params p, float* __restrict__ out)
{
    extern __shared__ float smem[];
    // fp32 pools
    float* sH  = smem;                  //  4096  current y
    float* sK1 = sH  + ROWS * HID;      //  4096
    float* sK2 = sK1 + ROWS * HID;      //  4096
    float* sK3 = sK2 + ROWS * HID;      //  4096
    float* sSt = sK3 + ROWS * HID;      //  4096  stage input (k4 dst aliases)
    float* sG  = sSt + ROWS * HID;      //  8192  gates
    // bf16 pools (16B-aligned: all offsets are multiples of 4096 floats)
    unsigned short* sHB  = (unsigned short*)(sG + ROWS * 2 * HID); // 4096 u16
    unsigned short* sStB = sHB  + ROWS * HID;                      // 4096 u16
    unsigned short* sAb  = sStB + ROWS * HID;                      // 4096 u16
    unsigned short* sCb  = sAb  + ROWS * HID;                      // 8192 u16
    const int row0 = blockIdx.x * ROWS;
    const int NE   = ROWS * HID;

    for (int i = threadIdx.x; i < NE; i += THREADS) {
        const float v = h0[(size_t)row0 * HID + i];
        sH[i]  = v;
        sHB[i] = f2bf(v);
    }
    __syncthreads();

    const float third = 1.0f / 3.0f;
    for (int b = 0; b < BATCH; ++b) {
        const float t0 = tsp[2 * b], t1 = tsp[2 * b + 1];
        const float dt = (t1 - t0) / (float)NSTEPS;
        for (int s = 0; s < NSTEPS; ++s) {
            const float t = t0 + (float)s * dt;

            dynamics(t, sH, sHB, sK1, sG, sAb, sCb, p);             // k1
            for (int i = threadIdx.x; i < NE; i += THREADS) {
                const float v = sH[i] + dt * third * sK1[i];
                sSt[i] = v;  sStB[i] = f2bf(v);
            }
            __syncthreads();

            dynamics(t + dt * third, sSt, sStB, sK2, sG, sAb, sCb, p);   // k2
            for (int i = threadIdx.x; i < NE; i += THREADS) {
                const float v = sH[i] + dt * (sK2[i] - third * sK1[i]);
                sSt[i] = v;  sStB[i] = f2bf(v);
            }
            __syncthreads();

            dynamics(t + 2.0f * dt * third, sSt, sStB, sK3, sG, sAb, sCb, p); // k3
            for (int i = threadIdx.x; i < NE; i += THREADS) {
                const float v = sH[i] + dt * (sK1[i] - sK2[i] + sK3[i]);
                sSt[i] = v;  sStB[i] = f2bf(v);
            }
            __syncthreads();

            dynamics(t + dt, sSt, sStB, sSt, sG, sAb, sCb, p);      // k4 -> sSt
            for (int i = threadIdx.x; i < NE; i += THREADS) {
                const float v = sH[i]
                    + dt * 0.125f * (sK1[i] + 3.0f * (sK2[i] + sK3[i]) + sSt[i]);
                sH[i]  = v;  sHB[i] = f2bf(v);
            }
            __syncthreads();
        }
        for (int i = threadIdx.x; i < NE; i += THREADS)
            out[(size_t)b * NROWS * HID + (size_t)row0 * HID + i] = sH[i];
        __syncthreads();
    }
}

__global__ void cvt_bf16_kernel(const float* __restrict__ src,
                                unsigned short* __restrict__ dst, int n)
{
    const int i = blockIdx.x * 256 + threadIdx.x;
    if (i < n) dst[i] = f2bf(src[i]);
}

extern "C" void kernel_launch(void* const* d_in, const int* in_sizes, int n_in,
                              void* d_out, int out_size, void* d_ws, size_t ws_size,
                              hipStream_t stream)
{
    (void)in_sizes; (void)n_in; (void)out_size; (void)ws_size;
    const float* h0    = (const float*)d_in[0];
    const float* tsp   = (const float*)d_in[1];
    /* d_in[2] adj_matrices: unused (spectral_reg is None) */
    const float* Wg_w  = (const float*)d_in[3];
    const float* Wg_b  = (const float*)d_in[4];
    const float* Wc_w  = (const float*)d_in[5];
    const float* Wc_b  = (const float*)d_in[6];
    const float* Wt_w  = (const float*)d_in[7];
    const float* Ws1_w = (const float*)d_in[8];
    const float* Ws1_b = (const float*)d_in[9];
    const float* Ws2_w = (const float*)d_in[10];
    const float* Ws2_b = (const float*)d_in[11];
    const float* Wf_w  = (const float*)d_in[12];
    const float* Wf_b  = (const float*)d_in[13];

    // bf16 weight pool in workspace (0.875 MB, L2-resident)
    unsigned short* wg = (unsigned short*)d_ws;        // 512*256
    unsigned short* wc = wg + 512 * 256;               // 256*256
    unsigned short* w1 = wc + 256 * 256;               // 256*256
    unsigned short* w2 = w1 + 256 * 256;               // 256*256
    unsigned short* wf = w2 + 256 * 256;               // 256*512

    cvt_bf16_kernel<<<(512 * 256 + 255) / 256, 256, 0, stream>>>(Wg_w,  wg, 512 * 256);
    cvt_bf16_kernel<<<(256 * 256 + 255) / 256, 256, 0, stream>>>(Wc_w,  wc, 256 * 256);
    cvt_bf16_kernel<<<(256 * 256 + 255) / 256, 256, 0, stream>>>(Ws1_w, w1, 256 * 256);
    cvt_bf16_kernel<<<(256 * 256 + 255) / 256, 256, 0, stream>>>(Ws2_w, w2, 256 * 256);
    cvt_bf16_kernel<<<(256 * 512 + 255) / 256, 256, 0, stream>>>(Wf_w,  wf, 256 * 512);

    Params p;
    p.Wg = wg; p.Wc = wc; p.Ws1 = w1; p.Ws2 = w2; p.Wf = wf;
    p.bg = Wg_b; p.bc = Wc_b; p.wt = Wt_w; p.b1 = Ws1_b; p.b2 = Ws2_b; p.bf = Wf_b;

    // LDS: 6x4096 + 8192 fp32  +  (4096*3 + 8192) u16  = 168 KB
    const size_t lds_bytes = (size_t)(6 * ROWS * HID + ROWS * 2 * HID) * sizeof(float)
                           + (size_t)(3 * ROWS * HID + ROWS * 2 * HID) * sizeof(unsigned short);
    (void)hipFuncSetAttribute((const void*)stode_kernel,
                              hipFuncAttributeMaxDynamicSharedMemorySize,
                              (int)lds_bytes);
    stode_kernel<<<NROWS / ROWS, THREADS, lds_bytes, stream>>>(
        h0, tsp, p, (float*)d_out);
}
